// BC_5274219839877
// MI455X (gfx1250) — matrix operand order repaired and checked by reading, not compiled, not run on hardware
//
#include <hip/hip_runtime.h>

// ---------------------------------------------------------------------------
// MI455X (gfx1250) implementation.
// Roofline: ~129 GFLOP vs ~121 MB of compulsory HBM traffic -> memory bound
// (~5us at 23.3 TB/s). bf16 WMMA (v_wmma_f32_16x16x32_bf16) gives far more
// math throughput than needed, so operands are converted f32->bf16 in
// registers and all reuse is served from the 192MB L2 (weights = 25MB
// resident, bf16 intermediates = 63MB resident in workspace).
// ---------------------------------------------------------------------------

typedef __attribute__((ext_vector_type(16))) __bf16 v16bf;
typedef __attribute__((ext_vector_type(8)))  float  v8f;

#define IMG_FEAT  2048
#define HIDDEN    1024
#define BA_HIDDEN 3072
#define GLIMPSE   8
#define BATCH     64
#define NV        128
#define NQ        32

// ---------------- weight-norm scale: sum-of-squares reduction --------------

__global__ void bc_init_meta(float* meta) {
    if (threadIdx.x == 0) { meta[0] = 0.f; meta[1] = 0.f; }
}

__global__ __launch_bounds__(256)
void bc_sumsq(const float* __restrict__ x, int n, float* __restrict__ accum) {
    __shared__ float red[256];
    float s = 0.f;
    for (int i = blockIdx.x * blockDim.x + threadIdx.x; i < n;
         i += gridDim.x * blockDim.x) {
        float v = x[i];
        s += v * v;
    }
    red[threadIdx.x] = s;
    __syncthreads();
    for (int off = 128; off > 0; off >>= 1) {
        if (threadIdx.x < off) red[threadIdx.x] += red[threadIdx.x + off];
        __syncthreads();
    }
    if (threadIdx.x == 0) atomicAdd(accum, red[0]);
}

__global__ void bc_finalize_scales(float* meta,
                                   const float* __restrict__ v_g,
                                   const float* __restrict__ q_g) {
    if (threadIdx.x == 0) {
        meta[2] = v_g[0] / sqrtf(meta[0]);  // scale for v path
        meta[3] = q_g[0] / sqrtf(meta[1]);  // scale for q path
    }
}

// ---------------- WMMA fragment loaders (wave32 ISA layouts) ---------------

// A 16x32 bf16 fragment: lane holds row (lane&15); lanes<16: K {0..7,16..23},
// lanes>=16: K {8..15,24..31}. Source is f32, K-contiguous (row-major, ld=K).
__device__ inline v16bf load_a_frag_f32(const float* __restrict__ row_ptr,
                                        int kb /* = kk + hi*8 */) {
    v16bf a;
    const float* p = row_ptr + kb;
#pragma unroll
    for (int j = 0; j < 8; ++j) a[j] = (__bf16)p[j];
#pragma unroll
    for (int j = 0; j < 8; ++j) a[8 + j] = (__bf16)p[16 + j];
    return a;
}

// B 32x16 bf16 fragment: lane holds column (lane&15); lanes<16: K {0..7,16..23},
// lanes>=16: K {8..15,24..31}. Source is f32, K-contiguous per output column (row of W).
__device__ inline v16bf load_b_frag_f32(const float* __restrict__ col_ptr,
                                        int kb /* = kk + hi*8 */) {
    v16bf b;
    const float* p = col_ptr + kb;
#pragma unroll
    for (int j = 0; j < 8; ++j) b[j] = (__bf16)p[j];
#pragma unroll
    for (int j = 0; j < 8; ++j) b[8 + j] = (__bf16)p[16 + j];
    return b;
}

__device__ inline v16bf load_b_frag_bf16(const __bf16* __restrict__ col_ptr,
                                         int kb) {
    v16bf b;
    const __bf16* p = col_ptr + kb;
#pragma unroll
    for (int j = 0; j < 8; ++j) b[j] = p[j];
#pragma unroll
    for (int j = 0; j < 8; ++j) b[8 + j] = p[16 + j];
    return b;
}

// ---------------- Stage 1/2: weight-normed GEMM + ReLU -> bf16 -------------
// Y[M x 3072] = bf16(relu(scale * X[M x K] @ W[3072 x K]^T + bias))
// Block tile 128(M) x 64(N); 8 waves; each wave does a 32x32 patch (2x2 WMMA).

__global__ __launch_bounds__(256)
void bc_wn_gemm_relu(const float* __restrict__ X,
                     const float* __restrict__ W,
                     const float* __restrict__ bias,
                     const float* __restrict__ scale_p,
                     __bf16* __restrict__ Y,
                     int K) {
    const int lane = threadIdx.x & 31;
    const int wave = threadIdx.x >> 5;
    const int wm   = wave & 3;   // 4 M sub-tiles
    const int wn   = wave >> 2;  // 2 N sub-tiles
    const int m0   = blockIdx.y * 128 + wm * 32;
    const int n0   = blockIdx.x * 64  + wn * 32;
    const int r    = lane & 15;
    const int hi   = lane >> 4;

    v8f acc[2][2] = {};

    for (int kk = 0; kk < K; kk += 32) {
        v16bf a[2], b[2];
#pragma unroll
        for (int mi = 0; mi < 2; ++mi)
            a[mi] = load_a_frag_f32(X + (size_t)(m0 + mi * 16 + r) * K,
                                    kk + hi * 8);
#pragma unroll
        for (int ni = 0; ni < 2; ++ni)
            b[ni] = load_b_frag_f32(W + (size_t)(n0 + ni * 16 + r) * K,
                                    kk + hi * 8);
#pragma unroll
        for (int mi = 0; mi < 2; ++mi)
#pragma unroll
            for (int ni = 0; ni < 2; ++ni)
                acc[mi][ni] = __builtin_amdgcn_wmma_f32_16x16x32_bf16(
                    false, a[mi], false, b[ni], (short)0, acc[mi][ni],
                    false, false);
    }

    const float scale = scale_p[0];
#pragma unroll
    for (int ni = 0; ni < 2; ++ni) {
        const int   col  = n0 + ni * 16 + r;
        const float bcol = bias[col];
#pragma unroll
        for (int mi = 0; mi < 2; ++mi) {
#pragma unroll
            for (int j = 0; j < 8; ++j) {
                const int row = m0 + mi * 16 + j + hi * 8;
                float val = scale * acc[mi][ni][j] + bcol;
                val = val > 0.f ? val : 0.f;
                Y[(size_t)row * BA_HIDDEN + col] = (__bf16)val;
            }
        }
    }
}

// ---------------- Stage 3: bhvq trilinear via WMMA -------------------------
// Per (b,h): logits = (v_[b] * hm[h]) @ q_[b]^T + h_bias[h]
// M=128, N=32, K=3072. One workgroup per (h,b); 8 waves, 16 rows each.

__global__ __launch_bounds__(256)
void bc_glimpse_logits(const __bf16* __restrict__ Vp,  // (B*NV) x 3072
                       const __bf16* __restrict__ Qp,  // (B*NQ) x 3072
                       const float* __restrict__ hm,   // 8 x 3072
                       const float* __restrict__ hbias,// 8
                       float* __restrict__ out) {      // B x 8 x NV x NQ
    const int h    = blockIdx.x;
    const int b    = blockIdx.y;
    const int lane = threadIdx.x & 31;
    const int wave = threadIdx.x >> 5;  // m tile index, 8 * 16 = 128 rows
    const int r    = lane & 15;
    const int hi   = lane >> 4;
    const int m0   = wave * 16;

    const __bf16* Vb  = Vp + (size_t)(b * NV) * BA_HIDDEN;
    const __bf16* Qb  = Qp + (size_t)(b * NQ) * BA_HIDDEN;
    const float*  hmr = hm + h * BA_HIDDEN;

    v8f acc[2] = {};

    for (int kk = 0; kk < BA_HIDDEN; kk += 32) {
        const int kba = kk + hi * 8;
        // A fragment: v_ row scaled elementwise by hm[h] (fused).
        v16bf a;
        const __bf16* pv = Vb + (size_t)(m0 + r) * BA_HIDDEN + kba;
        const float*  ph = hmr + kba;
#pragma unroll
        for (int j = 0; j < 8; ++j)
            a[j] = (__bf16)((float)pv[j] * ph[j]);
#pragma unroll
        for (int j = 0; j < 8; ++j)
            a[8 + j] = (__bf16)((float)pv[16 + j] * ph[16 + j]);

        v16bf bf[2];
#pragma unroll
        for (int ni = 0; ni < 2; ++ni)
            bf[ni] = load_b_frag_bf16(Qb + (size_t)(ni * 16 + r) * BA_HIDDEN,
                                      kk + hi * 8);
#pragma unroll
        for (int ni = 0; ni < 2; ++ni)
            acc[ni] = __builtin_amdgcn_wmma_f32_16x16x32_bf16(
                false, a, false, bf[ni], (short)0, acc[ni], false, false);
    }

    const float hb = hbias[h];
    float* ob = out + (((size_t)b * GLIMPSE + h) * NV) * NQ;
#pragma unroll
    for (int ni = 0; ni < 2; ++ni) {
        const int col = ni * 16 + r;
#pragma unroll
        for (int j = 0; j < 8; ++j) {
            const int row = m0 + j + hi * 8;
            ob[(size_t)row * NQ + col] = acc[ni][j] + hb;
        }
    }
}

// ---------------------------------------------------------------------------

extern "C" void kernel_launch(void* const* d_in, const int* in_sizes, int n_in,
                              void* d_out, int out_size, void* d_ws, size_t ws_size,
                              hipStream_t stream) {
    (void)in_sizes; (void)n_in; (void)out_size; (void)ws_size;
    const float* v      = (const float*)d_in[0];
    const float* q      = (const float*)d_in[1];
    const float* v_w    = (const float*)d_in[2];
    const float* v_g    = (const float*)d_in[3];
    const float* v_b    = (const float*)d_in[4];
    const float* q_w    = (const float*)d_in[5];
    const float* q_g    = (const float*)d_in[6];
    const float* q_b    = (const float*)d_in[7];
    const float* h_mat  = (const float*)d_in[8];
    const float* h_bias = (const float*)d_in[9];
    float* out = (float*)d_out;

    // Workspace layout: [0,256) meta floats; then bf16 v_ (50.3MB), bf16 q_.
    float*  meta = (float*)d_ws;
    __bf16* v_ws = (__bf16*)((char*)d_ws + 256);
    __bf16* q_ws = v_ws + (size_t)BATCH * NV * BA_HIDDEN;

    bc_init_meta<<<1, 32, 0, stream>>>(meta);
    bc_sumsq<<<256, 256, 0, stream>>>(v_w, BA_HIDDEN * IMG_FEAT, meta + 0);
    bc_sumsq<<<256, 256, 0, stream>>>(q_w, BA_HIDDEN * HIDDEN,   meta + 1);
    bc_finalize_scales<<<1, 32, 0, stream>>>(meta, v_g, q_g);

    // v_: M=8192, K=2048, N=3072 -> grid (3072/64, 8192/128)
    bc_wn_gemm_relu<<<dim3(48, 64), 256, 0, stream>>>(
        v, v_w, v_b, meta + 2, v_ws, IMG_FEAT);
    // q_: M=2048, K=1024, N=3072 -> grid (3072/64, 2048/128)
    bc_wn_gemm_relu<<<dim3(48, 16), 256, 0, stream>>>(
        q, q_w, q_b, meta + 3, q_ws, HIDDEN);

    // logits: one block per (h, b)
    bc_glimpse_logits<<<dim3(GLIMPSE, BATCH), 256, 0, stream>>>(
        v_ws, q_ws, h_mat, h_bias, out);
}
